// TRoPECrossAttention_41867341201565
// MI455X (gfx1250) — matrix operand order repaired and checked by reading, not compiled, not run on hardware
//
#include <hip/hip_runtime.h>

#define DEV __device__ __forceinline__

typedef __attribute__((ext_vector_type(16))) _Float16 v16h;
typedef __attribute__((ext_vector_type(8)))  _Float16 v8h;
typedef __attribute__((ext_vector_type(8)))  float    v8f;
typedef __attribute__((ext_vector_type(4)))  int      v4i;

constexpr int BATCH = 2, SEQ = 2048, DIMC = 1024, HEADS = 16, DHEAD = 64;
constexpr int INNER = HEADS * DHEAD;          // 1024
constexpr int NKV   = 3 * INNER;              // 3072 (q | k | v)
constexpr int ROWS  = BATCH * SEQ;            // 4096

#ifndef __has_builtin
#define __has_builtin(x) 0
#endif
#if __has_builtin(__builtin_amdgcn_global_load_async_to_lds_b128) && \
    __has_builtin(__builtin_amdgcn_s_wait_asynccnt)
#define ASYNC_COPY 1
#else
#define ASYNC_COPY 0
#endif

// 16-byte global -> LDS copy; async (ASYNCcnt-tracked, no VGPR bounce) when available.
DEV void cp16(_Float16* l, const _Float16* g) {
#if ASYNC_COPY
    __builtin_amdgcn_global_load_async_to_lds_b128((v4i*)g, (v4i*)l, 0, 0);
#else
    *(v8h*)l = *(const v8h*)g;
#endif
}
DEV void cp_wait() {
#if ASYNC_COPY
    __builtin_amdgcn_s_wait_asynccnt(0);
#endif
}

DEV v16h make_v16(v8h lo, v8h hi) {
    v16h r;
#pragma unroll
    for (int i = 0; i < 8; ++i) { r[i] = lo[i]; r[8 + i] = hi[i]; }
    return r;
}

// ---------------------------------------------------------------- packing kernels
__global__ void f32_to_f16_kernel(const float* __restrict__ s, _Float16* __restrict__ d, int n) {
    int i = blockIdx.x * blockDim.x + threadIdx.x;
    if (i < n) d[i] = (_Float16)s[i];
}

// W_q | W_kv concatenated and TRANSPOSED: dst[j][k] (j in [0,3072), k in [0,1024))
__global__ void pack_wqkv_t_kernel(const float* __restrict__ wq, const float* __restrict__ wkv,
                                   _Float16* __restrict__ dst) {
    int i = blockIdx.x * blockDim.x + threadIdx.x;
    if (i >= NKV * DIMC) return;
    int j = i / DIMC, k = i - j * DIMC;
    float v = (j < INNER) ? wq[(size_t)k * INNER + j]
                          : wkv[(size_t)k * (2 * INNER) + (j - INNER)];
    dst[i] = (_Float16)v;
}

// dst[c][r] = src[r][c]  (f32 [R,C] -> f16 [C,R])
__global__ void transpose_f32_to_f16_kernel(const float* __restrict__ src,
                                            _Float16* __restrict__ dst, int R, int C) {
    int i = blockIdx.x * blockDim.x + threadIdx.x;
    if (i >= R * C) return;
    int c = i / R, r = i - c * R;
    dst[i] = (_Float16)src[(size_t)r * C + c];
}

// ---------------------------------------------------------------- WMMA GEMM
// C[M,N] = A[M,K] * Bt[N,K]^T  (both f16 row-major); fp32 out (+bias).
// 256 thr = 8 waves; tile 128x128x32, double-buffered async LDS staging.
template <bool BIAS>
__global__ __launch_bounds__(256, 2)
void wmma_gemm_kernel(const _Float16* __restrict__ A, const _Float16* __restrict__ Bt,
                      float* __restrict__ C, const float* __restrict__ bias,
                      int M, int N, int K) {
    constexpr int BM = 128, BN = 128, BK = 32, PAD = BK + 8;   // 80B stride: conflict-free
    __shared__ _Float16 Al[2][BM * PAD];
    __shared__ _Float16 Bl[2][BN * PAD];

    const int tid  = threadIdx.x;
    const int lane = tid & 31;
    const int wid  = tid >> 5;
    const int wm   = wid >> 1;        // 0..3
    const int wn   = wid & 1;         // 0..1
    const int gm0  = blockIdx.y * BM;
    const int gn0  = blockIdx.x * BN;

    const int lrow  = lane & 15;
    const int hi    = lane >> 4;
    const int abase = hi * 16;        // A-frag K offset (contiguous K layout)
    const int bbase = hi * 16;        // B-frag K offset

    v8f acc[2][4];
#pragma unroll
    for (int m = 0; m < 2; ++m)
#pragma unroll
        for (int n = 0; n < 4; ++n) acc[m][n] = (v8f){};

    auto stage = [&](int buf, int k0) {
#pragma unroll
        for (int i = 0; i < 2; ++i) {
            int c = tid + i * 256;                 // 512 chunks of 8 f16 per matrix
            int r = c >> 2, kc = (c & 3) * 8;
            cp16(&Al[buf][r * PAD + kc], A  + (size_t)(gm0 + r) * K + k0 + kc);
            cp16(&Bl[buf][r * PAD + kc], Bt + (size_t)(gn0 + r) * K + k0 + kc);
        }
    };

    stage(0, 0);
    cp_wait();
    __syncthreads();

    int cur = 0;
    for (int k0 = 0; k0 < K; k0 += BK) {
        if (k0 + BK < K) stage(cur ^ 1, k0 + BK);   // prefetch next tile (async)

        v16h af[2], bf[4];
#pragma unroll
        for (int mt = 0; mt < 2; ++mt) {
            const _Float16* p = &Al[cur][(wm * 32 + mt * 16 + lrow) * PAD];
            af[mt] = make_v16(*(const v8h*)(p + abase), *(const v8h*)(p + 8 + abase));
        }
#pragma unroll
        for (int nt = 0; nt < 4; ++nt) {
            const _Float16* p = &Bl[cur][(wn * 64 + nt * 16 + lrow) * PAD + bbase];
            bf[nt] = make_v16(*(const v8h*)p, *(const v8h*)(p + 8));
        }
#pragma unroll
        for (int mt = 0; mt < 2; ++mt)
#pragma unroll
            for (int nt = 0; nt < 4; ++nt)
                acc[mt][nt] = __builtin_amdgcn_wmma_f32_16x16x32_f16(
                    false, af[mt], false, bf[nt], (short)0, acc[mt][nt], false, false);

        cp_wait();
        __syncthreads();
        cur ^= 1;
    }

#pragma unroll
    for (int mt = 0; mt < 2; ++mt) {
#pragma unroll
        for (int nt = 0; nt < 4; ++nt) {
            int row0 = gm0 + wm * 32 + mt * 16 + hi * 8;
            int col  = gn0 + wn * 64 + nt * 16 + lrow;
            float bv = BIAS ? bias[col] : 0.0f;
#pragma unroll
            for (int r = 0; r < 8; ++r)
                C[(size_t)(row0 + r) * N + col] = acc[mt][nt][r] + bv;
        }
    }
}

// ---------------------------------------------------------------- RoPE (q,k) + pack heads
// qkv32: [b, n, 3072] fp32.  Out: q/k f16 as [b*H, n, 64] (roped).
__global__ void rope_pack_kernel(const float* __restrict__ qkv,
                                 const int* __restrict__ tq, const int* __restrict__ tk,
                                 _Float16* __restrict__ qh, _Float16* __restrict__ kh) {
    int tid = blockIdx.x * blockDim.x + threadIdx.x;     // ROWS*HEADS*32 pairs
    if (tid >= ROWS * HEADS * (DHEAD / 2)) return;
    int p  = tid & 31;
    int h  = (tid >> 5) & (HEADS - 1);
    int bn = tid >> 9;
    int b  = bn >> 11;
    int n  = bn & (SEQ - 1);

    const float* row = qkv + (size_t)bn * NKV;
    float inv = __expf(-(float)(2 * p) * (9.210340371976184f / (float)DHEAD));
    float fq = (float)tq[bn] * inv;
    float fk = (float)tk[bn] * inv;
    float sq, cq, sk, ck;
    __sincosf(fq, &sq, &cq);
    __sincosf(fk, &sk, &ck);

    int d0 = h * DHEAD + 2 * p;
    float q1 = row[d0],         q2 = row[d0 + 1];
    float k1 = row[INNER + d0], k2 = row[INNER + d0 + 1];

    size_t o = ((size_t)(b * HEADS + h) * SEQ + n) * DHEAD + 2 * p;
    qh[o]     = (_Float16)(q1 * cq - q2 * sq);
    qh[o + 1] = (_Float16)(q1 * sq + q2 * cq);
    kh[o]     = (_Float16)(k1 * ck - k2 * sk);
    kh[o + 1] = (_Float16)(k1 * sk + k2 * ck);
}

// v packed TRANSPOSED per head: vt[bh][d][n]  (coalesced writes)
__global__ void v_pack_t_kernel(const float* __restrict__ qkv, _Float16* __restrict__ vt) {
    int i = blockIdx.x * blockDim.x + threadIdx.x;       // [bh][d][n], 4M
    if (i >= ROWS * INNER) return;
    int n  = i & (SEQ - 1);
    int d  = (i >> 11) & (DHEAD - 1);
    int bh = i >> 17;                                    // 64*2048 = 2^17
    int b = bh >> 4, h = bh & (HEADS - 1);
    float v = qkv[((size_t)b * SEQ + n) * NKV + 2 * INNER + h * DHEAD + d];
    vt[i] = (_Float16)v;
}

// ---------------------------------------------------------------- flash attention
// grid: (B*H, SEQ/64), block 128 (4 waves). Wave = 16 query rows; 64-key blocks.
__global__ __launch_bounds__(128, 2)
void flash_attn_kernel(const _Float16* __restrict__ qh, const _Float16* __restrict__ kh,
                       const _Float16* __restrict__ vt, _Float16* __restrict__ oh) {
    constexpr int KB = 64, KPAD = DHEAD + 8, VPAD = KB + 8, PPAD = KB + 8;  // 144B strides
    __shared__ _Float16 Kl[KB * KPAD];                 // [key][d]
    __shared__ _Float16 Vl[DHEAD * VPAD];              // [d][key]
    __shared__ _Float16 Pl[4][16 * PPAD];              // per-wave P scratch [qrow][key]

    const int tid  = threadIdx.x;
    const int lane = tid & 31;
    const int w    = tid >> 5;
    const int bh   = blockIdx.x;
    const int q0   = blockIdx.y * 64 + w * 16;
    const int lrow  = lane & 15;
    const int hi    = lane >> 4;
    const int abase = hi * 16;
    const int bbase = hi * 16;

    // Q fragments (A-operand; head dim = K split 2x32)
    v16h qa[2];
    const _Float16* qp = qh + ((size_t)bh * SEQ + q0 + lrow) * DHEAD;
#pragma unroll
    for (int c = 0; c < 2; ++c)
        qa[c] = make_v16(*(const v8h*)(qp + c * 32 + abase),
                         *(const v8h*)(qp + c * 32 + 8 + abase));

    float mr[8], lr[8];
    v8f o[4];
#pragma unroll
    for (int r = 0; r < 8; ++r) { mr[r] = -1e30f; lr[r] = 0.0f; }
#pragma unroll
    for (int nt = 0; nt < 4; ++nt) o[nt] = (v8f){};

    const _Float16* kp  = kh + (size_t)bh * SEQ * DHEAD;
    const _Float16* vtp = vt + (size_t)bh * DHEAD * SEQ;
    const float scale = 0.125f;                        // 64^-0.5

    for (int kb = 0; kb < SEQ; kb += KB) {
        // async-stage K [64 keys x 64 d] and V^T [64 d x 64 keys]
#pragma unroll
        for (int i = 0; i < 4; ++i) {
            int c = tid + i * 128;                     // 512 chunks of 8 f16 each matrix
            int r = c >> 3, x0 = (c & 7) * 8;
            cp16(&Kl[r * KPAD + x0], kp  + (size_t)(kb + r) * DHEAD + x0);
            cp16(&Vl[r * VPAD + x0], vtp + (size_t)r * SEQ + kb + x0);
        }
        cp_wait();
        __syncthreads();

        // S = Q K^T : 4 key-tiles x (K=64 over 2 chunks)
        v8f s[4];
#pragma unroll
        for (int nt = 0; nt < 4; ++nt) {
            v8f c0 = (v8f){};
#pragma unroll
            for (int ch = 0; ch < 2; ++ch) {
                const _Float16* p = &Kl[(nt * 16 + lrow) * KPAD + ch * 32 + bbase];
                v16h bf = make_v16(*(const v8h*)p, *(const v8h*)(p + 8));
                c0 = __builtin_amdgcn_wmma_f32_16x16x32_f16(false, qa[ch], false, bf,
                                                            (short)0, c0, false, false);
            }
#pragma unroll
            for (int r = 0; r < 8; ++r) c0[r] *= scale;
            s[nt] = c0;
        }

        // online softmax (row = r + 8*hi, spread over 16 lanes)
        float mn[8], alpha[8];
#pragma unroll
        for (int r = 0; r < 8; ++r) {
            float t = fmaxf(fmaxf(s[0][r], s[1][r]), fmaxf(s[2][r], s[3][r]));
            t = fmaxf(t, __shfl_xor(t, 1));
            t = fmaxf(t, __shfl_xor(t, 2));
            t = fmaxf(t, __shfl_xor(t, 4));
            t = fmaxf(t, __shfl_xor(t, 8));
            mn[r] = fmaxf(mr[r], t);
            alpha[r] = __expf(mr[r] - mn[r]);
            mr[r] = mn[r];
        }
#pragma unroll
        for (int r = 0; r < 8; ++r) {
            float t = 0.0f;
#pragma unroll
            for (int nt = 0; nt < 4; ++nt) {
                s[nt][r] = __expf(s[nt][r] - mn[r]);
                t += s[nt][r];
            }
            t += __shfl_xor(t, 1);
            t += __shfl_xor(t, 2);
            t += __shfl_xor(t, 4);
            t += __shfl_xor(t, 8);
            lr[r] = lr[r] * alpha[r] + t;
        }
#pragma unroll
        for (int nt = 0; nt < 4; ++nt)
#pragma unroll
            for (int r = 0; r < 8; ++r) o[nt][r] *= alpha[r];

        // P: C-layout -> per-wave LDS scratch -> A-layout
        _Float16* pw = &Pl[w][0];
#pragma unroll
        for (int r = 0; r < 8; ++r) {
            int prow = r + hi * 8;
#pragma unroll
            for (int nt = 0; nt < 4; ++nt)
                pw[prow * PPAD + nt * 16 + lrow] = (_Float16)s[nt][r];
        }
        __builtin_amdgcn_wave_barrier();
        const _Float16* pr = &Pl[w][lrow * PPAD];
        v16h pf[2];
#pragma unroll
        for (int c = 0; c < 2; ++c)
            pf[c] = make_v16(*(const v8h*)(pr + c * 32 + abase),
                             *(const v8h*)(pr + c * 32 + 8 + abase));

        // O += P V : 4 d-tiles x 2 key-chunks
#pragma unroll
        for (int nt = 0; nt < 4; ++nt) {
#pragma unroll
            for (int c = 0; c < 2; ++c) {
                const _Float16* bp = &Vl[(nt * 16 + lrow) * VPAD + c * 32 + bbase];
                v16h bf = make_v16(*(const v8h*)bp, *(const v8h*)(bp + 8));
                o[nt] = __builtin_amdgcn_wmma_f32_16x16x32_f16(false, pf[c], false, bf,
                                                               (short)0, o[nt], false, false);
            }
        }
        __syncthreads();
    }

    // O normalized; heads re-interleaved: oh[b, n, h*64 + d] f16
    int b = bh / HEADS, h = bh % HEADS;
#pragma unroll
    for (int nt = 0; nt < 4; ++nt) {
        int col = h * DHEAD + nt * 16 + lrow;
#pragma unroll
        for (int r = 0; r < 8; ++r) {
            int row = q0 + r + hi * 8;
            oh[((size_t)b * SEQ + row) * INNER + col] = (_Float16)(o[nt][r] / lr[r]);
        }
    }
}

// ---------------------------------------------------------------- launcher
extern "C" void kernel_launch(void* const* d_in, const int* in_sizes, int n_in,
                              void* d_out, int out_size, void* d_ws, size_t ws_size,
                              hipStream_t stream) {
    const float* x     = (const float*)d_in[0];
    const int*   tq    = (const int*)d_in[1];
    const int*   tk    = (const int*)d_in[2];
    const float* W_q   = (const float*)d_in[3];
    const float* W_kv  = (const float*)d_in[4];
    const float* W_out = (const float*)d_in[5];
    const float* b_out = (const float*)d_in[6];
    float* out = (float*)d_out;

    size_t off = 0;
    auto carve = [&](size_t bytes) {
        void* p = (char*)d_ws + off;
        off = (off + bytes + 255) & ~(size_t)255;
        return p;
    };
    _Float16* xh     = (_Float16*)carve((size_t)ROWS * DIMC * 2);
    _Float16* wqkvt  = (_Float16*)carve((size_t)NKV * DIMC * 2);     // [3072][1024]
    _Float16* wot    = (_Float16*)carve((size_t)DIMC * INNER * 2);   // [1024][1024] (W_out^T)
    float*    qkv32  = (float*)carve((size_t)ROWS * NKV * 4);
    _Float16* qhp    = (_Float16*)carve((size_t)ROWS * INNER * 2);
    _Float16* khp    = (_Float16*)carve((size_t)ROWS * INNER * 2);
    _Float16* vtp    = (_Float16*)carve((size_t)ROWS * INNER * 2);   // [bh][d][n]
    _Float16* ohp    = (_Float16*)carve((size_t)ROWS * INNER * 2);

    // 1) pack/convert
    {
        int n = ROWS * DIMC;
        f32_to_f16_kernel<<<(n + 255) / 256, 256, 0, stream>>>(x, xh, n);
        int nw = NKV * DIMC;
        pack_wqkv_t_kernel<<<(nw + 255) / 256, 256, 0, stream>>>(W_q, W_kv, wqkvt);
        int no = DIMC * INNER;
        transpose_f32_to_f16_kernel<<<(no + 255) / 256, 256, 0, stream>>>(W_out, wot,
                                                                          INNER, DIMC);
    }
    // 2) QKV projection: [4096,1024] x [1024,3072] -> fp32
    {
        dim3 g(NKV / 128, ROWS / 128);
        wmma_gemm_kernel<false><<<g, 256, 0, stream>>>(xh, wqkvt, qkv32, nullptr,
                                                       ROWS, NKV, DIMC);
    }
    // 3) RoPE(q,k) + head packing; V packed transposed
    {
        int n = ROWS * HEADS * (DHEAD / 2);
        rope_pack_kernel<<<(n + 255) / 256, 256, 0, stream>>>(qkv32, tq, tk, qhp, khp);
        int nv = ROWS * INNER;
        v_pack_t_kernel<<<(nv + 255) / 256, 256, 0, stream>>>(qkv32, vtp);
    }
    // 4) flash attention
    {
        dim3 g(BATCH * HEADS, SEQ / 64);
        flash_attn_kernel<<<g, 128, 0, stream>>>(qhp, khp, vtp, ohp);
    }
    // 5) output projection + bias -> fp32 d_out
    {
        dim3 g(DIMC / 128, ROWS / 128);
        wmma_gemm_kernel<true><<<g, 256, 0, stream>>>(ohp, wot, out, b_out,
                                                      ROWS, DIMC, DIMC);
    }
}
